// GATEncoder_62062277427643
// MI455X (gfx1250) — matrix operand order repaired and checked by reading, not compiled, not run on hardware
//
#include <hip/hip_runtime.h>
#include <hip/hip_bf16.h>

// ---------------------------------------------------------------------------
// GAT encoder, 3 layers, H=4 heads x C=64, F=128 in, N=50000 nodes,
// E=800000 edges (+N self loops), B=256 graphs.
//
// Per layer (stream-ordered):
//   1. k_h2bf    : layer input f32 [N,K] -> bf16 [N,K] (RNE, uint4 stores)
//   2. k_wtrans  : W [K,256] f32 -> Wt [256,K] bf16 (transpose + RNE)
//   3. k_gemm_wmma : xh = h @ W; inner loop = 4x global_load_b128 + 1x
//                    v_wmma_f32_16x16x32_bf16 (f32 accum)
//   4. k_alpha   : alpha_s/alpha_d [N,4] head dots
//   5. k_init_nodes / k_init_out : seg-max = ord(-inf), z = 0, acc = bias
//   6. k_edge_logits_max : leakyrelu logits + ordered-u32 atomicMax seg-max
//   7. k_edge_exp_sum    : e = exp(l - m[dst]) in place + atomic z sum
//   8. k_edge_agg : wave32 per edge, acc[dst] += xh[src]*alpha (f32 atomics,
//                   lanes cover 32 consecutive dwords per step -> coalesced)
//   9. k_relu (layers 1,2)
// Then global max pool via ordered-u32 atomicMax over batch, decode to f32.
// ---------------------------------------------------------------------------

#define N_NODES 50000
#define N_EDGES 800000
#define EPLUS   (N_EDGES + N_NODES)   // edges incl. self loops = 850000
#define NBATCH  256
#define F_IN    128
#define NH      4
#define NC      64
#define HC      256                   // NH*NC

typedef __attribute__((ext_vector_type(16))) __bf16 v16bf;
typedef __attribute__((ext_vector_type(8)))  float  v8f;

static __device__ __forceinline__ unsigned f2o(float f) {
  unsigned b = __float_as_uint(f);
  return (b & 0x80000000u) ? ~b : (b | 0x80000000u);
}
static __device__ __forceinline__ float o2f(unsigned u) {
  unsigned b = (u & 0x80000000u) ? (u & 0x7fffffffu) : ~u;
  return __uint_as_float(b);
}
#define ORD_NEG_INF 0x007FFFFFu       // f2o(-inf)

// round-to-nearest-even f32 -> bf16 (bit level)
static __device__ __forceinline__ unsigned short bfr(float f) {
  unsigned b = __float_as_uint(f);
  b += 0x7fffu + ((b >> 16) & 1u);
  return (unsigned short)(b >> 16);
}
static __device__ __forceinline__ unsigned pk2(float lo, float hi) {
  return (unsigned)bfr(lo) | ((unsigned)bfr(hi) << 16);
}

union BfFrag { v16bf v; unsigned u[8]; uint4 q[2]; };

// --------------------------- kernels ---------------------------------------

// f32 [rows,K] row-major -> bf16 [rows,K] row-major; 8 elems per thread.
__global__ void k_h2bf(const float* __restrict__ h, unsigned short* __restrict__ xb,
                       int total8) {
  int tid = blockIdx.x * blockDim.x + threadIdx.x;
  if (tid >= total8) return;
  const float4* p = (const float4*)h + (size_t)tid * 2;
  float4 f0 = p[0], f1 = p[1];
  uint4 o;
  o.x = pk2(f0.x, f0.y);
  o.y = pk2(f0.z, f0.w);
  o.z = pk2(f1.x, f1.y);
  o.w = pk2(f1.z, f1.w);
  ((uint4*)xb)[tid] = o;
}

// Wt[n*K + k] = bf16(W[k*HC + n])
__global__ void k_wtrans(const float* __restrict__ W, unsigned short* __restrict__ Wt, int K) {
  int tid = blockIdx.x * blockDim.x + threadIdx.x;
  if (tid >= HC * K) return;
  int n = tid / K, k = tid - n * K;
  Wt[tid] = bfr(W[(size_t)k * HC + n]);
}

// xh[N,256] = A[N,K](bf16) @ W[K,256]; one wave per 16x16 tile.
// Inner loop: 2x b128 A loads + 2x b128 B loads + 1x v_wmma_f32_16x16x32_bf16.
__global__ void k_gemm_wmma(const unsigned short* __restrict__ xb,
                            const unsigned short* __restrict__ Wt,
                            float* __restrict__ xh, int K) {
  const int lane = threadIdx.x;        // 0..31
  const int ll = lane & 15;
  const int lh = lane >> 4;            // K-half selector per ISA layout
  const int m0 = blockIdx.x * 16;
  const int n0 = (blockIdx.y * 8 + threadIdx.y) * 16;

  const unsigned short* arow = xb + (size_t)(m0 + ll) * K;
  const unsigned short* brow = Wt + (size_t)(n0 + ll) * K;

  v8f c = {};
  for (int k = 0; k < K; k += 32) {
    // A 16x32 bf16: lanes 0-15 -> K k..k+7 (V0-3), k+16..k+23 (V4-7);
    //               lanes16-31 -> K k+8..k+15,      k+24..k+31.
    BfFrag a;
    const int ka = k + lh * 8;
    a.q[0] = *(const uint4*)(arow + ka);        // 8 bf16, 16B aligned
    a.q[1] = *(const uint4*)(arow + ka + 16);
    // B 32x16 bf16: lane holds column n0+ll; same K order as A:
    // lanes 0-15 K k..k+7, k+16..k+23; lanes 16-31 K k+8..k+15, k+24..k+31.
    BfFrag bb;
    bb.q[0] = *(const uint4*)(brow + ka);
    bb.q[1] = *(const uint4*)(brow + ka + 16);
    v16bf b = bb.v;
    c = __builtin_amdgcn_wmma_f32_16x16x32_bf16(
            /*neg_a=*/false, a.v, /*neg_b=*/false, b,
            /*c_mod=*/(short)0, c, /*reuse_a=*/false, /*reuse_b=*/false);
  }
  // C 16x16 f32: lane col = n0+ll, rows m0 + lh*8 + r
  float* orow = xh + (size_t)(m0 + lh * 8) * HC + n0 + ll;
#pragma unroll
  for (int r = 0; r < 8; ++r) orow[(size_t)r * HC] = c[r];
}

// alpha_s[n,h] = dot(xh[n, h*64:(h+1)*64], a_src[h]); same for alpha_d.
__global__ void k_alpha(const float* __restrict__ xh,
                        const float* __restrict__ as_, const float* __restrict__ ad_,
                        float* __restrict__ als, float* __restrict__ ald) {
  int tid = blockIdx.x * blockDim.x + threadIdx.x;
  if (tid >= N_NODES * NH) return;
  int n = tid >> 2, h = tid & 3;
  const float* xr = xh + (size_t)n * HC + h * NC;
  const float* a1 = as_ + h * NC;
  const float* a2 = ad_ + h * NC;
  float s1 = 0.f, s2 = 0.f;
#pragma unroll 8
  for (int cix = 0; cix < NC; ++cix) {
    float v = xr[cix];
    s1 += v * a1[cix];
    s2 += v * a2[cix];
  }
  als[tid] = s1;
  ald[tid] = s2;
}

__global__ void k_init_nodes(unsigned* __restrict__ maxb, float* __restrict__ z) {
  int tid = blockIdx.x * blockDim.x + threadIdx.x;
  if (tid >= N_NODES * NH) return;
  maxb[tid] = ORD_NEG_INF;
  z[tid] = 0.f;
}

__global__ void k_init_out(float* __restrict__ acc, const float* __restrict__ bias) {
  int tid = blockIdx.x * blockDim.x + threadIdx.x;
  if (tid >= N_NODES * HC) return;
  acc[tid] = bias[tid & (HC - 1)];
}

// logits + leakyrelu + segment max (ordered-uint atomicMax)
__global__ void k_edge_logits_max(const float* __restrict__ als, const float* __restrict__ ald,
                                  const int* __restrict__ ei,
                                  float* __restrict__ logits, unsigned* __restrict__ maxb) {
  int tid = blockIdx.x * blockDim.x + threadIdx.x;
  if (tid >= EPLUS * NH) return;
  int e = tid >> 2, h = tid & 3;
  int s, d;
  if (e < N_EDGES) { s = ei[e]; d = ei[N_EDGES + e]; }
  else             { s = d = e - N_EDGES; }
  float l = als[s * NH + h] + ald[d * NH + h];
  l = l > 0.f ? l : 0.2f * l;                 // LeakyReLU(0.2)
  logits[tid] = l;
  atomicMax(maxb + d * NH + h, f2o(l));
}

// e = exp(l - m[dst]) (in place over logits) + segment sum z
__global__ void k_edge_exp_sum(const int* __restrict__ ei,
                               const unsigned* __restrict__ maxb,
                               float* __restrict__ logits, float* __restrict__ z) {
  int tid = blockIdx.x * blockDim.x + threadIdx.x;
  if (tid >= EPLUS * NH) return;
  int e = tid >> 2, h = tid & 3;
  int d = (e < N_EDGES) ? ei[N_EDGES + e] : (e - N_EDGES);
  float m  = o2f(maxb[d * NH + h]);
  float ev = __expf(logits[tid] - m);
  logits[tid] = ev;
  unsafeAtomicAdd(z + d * NH + h, ev);
}

// wave32 per edge: acc[dst,:] += xh[src,:] * alpha[e,head]
__global__ void k_edge_agg(const float* __restrict__ xh,
                           const float* __restrict__ ev, const float* __restrict__ z,
                           const int* __restrict__ ei, float* __restrict__ acc) {
  int wid  = blockIdx.x * 8 + (threadIdx.x >> 5);   // edge id
  int lane = threadIdx.x & 31;
  if (wid >= EPLUS) return;
  int s, d;
  if (wid < N_EDGES) { s = ei[wid]; d = ei[N_EDGES + wid]; }
  else               { s = d = wid - N_EDGES; }
  const float* xr = xh + (size_t)s * HC;
  float* orow     = acc + (size_t)d * HC;
  __builtin_prefetch(xr + lane, 0, 0);              // global_prefetch_b8
  float wh[NH];
#pragma unroll
  for (int h = 0; h < NH; ++h)
    wh[h] = ev[(size_t)wid * NH + h] / (z[d * NH + h] + 1e-16f);
#pragma unroll
  for (int k = 0; k < 8; ++k) {
    int idx = lane + (k << 5);                      // head = k>>1
    unsafeAtomicAdd(orow + idx, xr[idx] * wh[k >> 1]);
  }
}

__global__ void k_relu(float* __restrict__ acc) {
  int tid = blockIdx.x * blockDim.x + threadIdx.x;
  if (tid >= N_NODES * HC) return;
  acc[tid] = fmaxf(acc[tid], 0.f);
}

__global__ void k_pool_init(unsigned* __restrict__ pu) {
  int tid = blockIdx.x * blockDim.x + threadIdx.x;
  if (tid >= NBATCH * HC) return;
  pu[tid] = ORD_NEG_INF;
}

__global__ void k_pool_max(const float* __restrict__ h, const int* __restrict__ batch,
                           unsigned* __restrict__ pu) {
  int tid = blockIdx.x * blockDim.x + threadIdx.x;
  if (tid >= N_NODES * HC) return;
  int n = tid >> 8, f = tid & (HC - 1);
  atomicMax(pu + batch[n] * HC + f, f2o(h[tid]));
}

__global__ void k_pool_decode(const unsigned* __restrict__ pu, float* __restrict__ out) {
  int tid = blockIdx.x * blockDim.x + threadIdx.x;
  if (tid >= NBATCH * HC) return;
  out[tid] = o2f(pu[tid]);
}

// --------------------------- launch ----------------------------------------

static inline int cdiv(long long a, long long b) { return (int)((a + b - 1) / b); }

extern "C" void kernel_launch(void* const* d_in, const int* in_sizes, int n_in,
                              void* d_out, int out_size, void* d_ws, size_t ws_size,
                              hipStream_t stream) {
  (void)in_sizes; (void)n_in; (void)out_size; (void)ws_size;

  const float* x     = (const float*)d_in[0];
  const int*   ei    = (const int*)d_in[1];
  const int*   batch = (const int*)d_in[2];
  const float* Ws[3]  = { (const float*)d_in[3],  (const float*)d_in[7],  (const float*)d_in[11] };
  const float* aSs[3] = { (const float*)d_in[4],  (const float*)d_in[8],  (const float*)d_in[12] };
  const float* aDs[3] = { (const float*)d_in[5],  (const float*)d_in[9],  (const float*)d_in[13] };
  const float* bs[3]  = { (const float*)d_in[6],  (const float*)d_in[10], (const float*)d_in[14] };
  const int    Ks[3]  = { F_IN, HC, HC };

  // workspace carve-up (all offsets 256B aligned; total ~145.2 MB)
  char* w = (char*)d_ws;
  float*          xh     = (float*)(w);                          // 50000*256 f32
  float*          acc    = (float*)(w + 51200000ull);            // 50000*256 f32
  float*          logits = (float*)(w + 102400000ull);           // 850000*4 f32 (reused as exp)
  float*          als    = (float*)(w + 116000000ull);           // 50000*4
  float*          ald    = (float*)(w + 116800000ull);           // 50000*4
  unsigned*       maxb   = (unsigned*)(w + 117600000ull);        // 50000*4
  float*          zbuf   = (float*)(w + 118400000ull);           // 50000*4
  unsigned short* Wt     = (unsigned short*)(w + 119200000ull);  // 256*256 bf16
  unsigned*       pooled = (unsigned*)(w + 119331072ull);        // 256*256 u32
  unsigned short* xb     = (unsigned short*)(w + 119593216ull);  // 50000*256 bf16

  const int T = 256;
  const float* hin = x;

  for (int L = 0; L < 3; ++L) {
    const int K = Ks[L];

    k_h2bf<<<cdiv((long long)N_NODES * K / 8, T), T, 0, stream>>>(hin, xb, N_NODES * K / 8);
    k_wtrans<<<cdiv((long long)HC * K, T), T, 0, stream>>>(Ws[L], Wt, K);

    dim3 gg(N_NODES / 16, HC / (16 * 8));   // 3125 x 2 blocks
    dim3 gb(32, 8);                         // 8 waves, one 16x16 tile each
    k_gemm_wmma<<<gg, gb, 0, stream>>>(xb, Wt, xh, K);

    k_alpha<<<cdiv((long long)N_NODES * NH, T), T, 0, stream>>>(xh, aSs[L], aDs[L], als, ald);
    k_init_nodes<<<cdiv((long long)N_NODES * NH, T), T, 0, stream>>>(maxb, zbuf);
    k_init_out<<<cdiv((long long)N_NODES * HC, T), T, 0, stream>>>(acc, bs[L]);

    k_edge_logits_max<<<cdiv((long long)EPLUS * NH, T), T, 0, stream>>>(als, ald, ei, logits, maxb);
    k_edge_exp_sum<<<cdiv((long long)EPLUS * NH, T), T, 0, stream>>>(ei, maxb, logits, zbuf);
    k_edge_agg<<<cdiv(EPLUS, 8), T, 0, stream>>>(xh, logits, zbuf, ei, acc);

    if (L < 2) k_relu<<<cdiv((long long)N_NODES * HC, T), T, 0, stream>>>(acc);
    hin = acc;   // acc fully consumed by next layer's k_h2bf before reinit
  }

  k_pool_init<<<cdiv((long long)NBATCH * HC, T), T, 0, stream>>>(pooled);
  k_pool_max<<<cdiv((long long)N_NODES * HC, T), T, 0, stream>>>(acc, batch, pooled);
  k_pool_decode<<<cdiv((long long)NBATCH * HC, T), T, 0, stream>>>(pooled, (float*)d_out);
}
